// CausalAttention_33758442947011
// MI455X (gfx1250) — matrix operand order repaired and checked by reading, not compiled, not run on hardware
//
#include <hip/hip_runtime.h>
#include <hip/hip_bf16.h>
#include <math.h>

typedef __attribute__((ext_vector_type(16))) __bf16 v16bf;
typedef __attribute__((ext_vector_type(8)))  __bf16 v8bf;
typedef __attribute__((ext_vector_type(8)))  float  v8f;

#define HDIM  1024
#define NH    16
#define HD    64
#define BATCH 2
#define SEQ   2048
#define NTOK  (BATCH * SEQ)   // 4096
#define LDSW  40              // padded LDS row stride (bf16 elems) -> conflict-free

// Async Global->LDS staging (ASYNCcnt path) when the toolchain exposes it.
#if defined(__gfx1250__) && __has_builtin(__builtin_amdgcn_global_load_async_to_lds_b128)
#define GEMM_ASYNC 1
#define AS1 __attribute__((address_space(1)))
#define AS3 __attribute__((address_space(3)))
typedef int v4i_vs __attribute__((vector_size(16)));   // matches builtin param type
#else
#define GEMM_ASYNC 0
#endif

// ---------------------------------------------------------------- helpers ---

__device__ __forceinline__ __bf16 f2bf(float f) {
  union { float f; unsigned u; } a; a.f = f;
  unsigned u = a.u;
  u += 0x7FFFu + ((u >> 16) & 1u);          // round-to-nearest-even
  union { unsigned short s; __bf16 b; } c; c.s = (unsigned short)(u >> 16);
  return c.b;
}

__device__ __forceinline__ v16bf make_frag(v8bf lo, v8bf hi) {
  v16bf r;
#pragma unroll
  for (int i = 0; i < 8; ++i) { r[i] = lo[i]; r[i + 8] = hi[i]; }
  return r;
}

__device__ __forceinline__ v8f zero8() {
  v8f z = {0.f, 0.f, 0.f, 0.f, 0.f, 0.f, 0.f, 0.f};
  return z;
}

__device__ __forceinline__ v8f wmma_bf16(v16bf a, v16bf b, v8f c) {
  // D = A(16x32 bf16) x B(32x16 bf16) + C(16x16 f32)
  return __builtin_amdgcn_wmma_f32_16x16x32_bf16(
      false, a, false, b, (short)0, c, false, false);
}

__device__ __forceinline__ void wait_async_all() {
#if GEMM_ASYNC
  asm volatile("s_wait_asynccnt 0x0" ::: "memory");
#endif
}

// ---------------------------------------------- precision conversion paths ---

__global__ void __launch_bounds__(256)
cast_bf16_kernel(const float* __restrict__ in, __bf16* __restrict__ out, int n) {
  int i = (blockIdx.x * blockDim.x + threadIdx.x) * 4;
  if (i < n) {
    float4 v = *(const float4*)(in + i);
    out[i + 0] = f2bf(v.x);
    out[i + 1] = f2bf(v.y);
    out[i + 2] = f2bf(v.z);
    out[i + 3] = f2bf(v.w);
  }
}

// in: [R][C] f32 (row-major)  ->  out: [C][R] bf16
__global__ void __launch_bounds__(256)
transpose_cast_kernel(const float* __restrict__ in, __bf16* __restrict__ out,
                      int R, int C) {
  __shared__ float tile[32][33];
  const int c0 = blockIdx.x * 32, r0 = blockIdx.y * 32;
  const int tx = threadIdx.x, ty = threadIdx.y;   // 32 x 8
#pragma unroll
  for (int i = 0; i < 32; i += 8)
    tile[ty + i][tx] = in[(size_t)(r0 + ty + i) * C + c0 + tx];
  __syncthreads();
#pragma unroll
  for (int i = 0; i < 32; i += 8)
    out[(size_t)(c0 + ty + i) * R + r0 + tx] = f2bf(tile[tx][ty + i]);
}

// -------------------------------------------------- 128x128 WMMA GEMM core ---
// C[M][N] = A[M][K](bf16,row-major) x Bt[N][K](bf16,row-major)^T
// block = 256 threads = 8 waves in a 2(M) x 4(N) grid; per-wave 64x32 of C.
// Double-buffered LDS; slab k+1 streams in (async-to-LDS when available)
// while slab k feeds the WMMA pipe.

template <typename Epi>
__device__ __forceinline__ void gemm_bf16_tile(const __bf16* __restrict__ A,
                                               const __bf16* __restrict__ Bt,
                                               int K, Epi epi) {
  __shared__ __bf16 As[2][128 * LDSW];
  __shared__ __bf16 Bs[2][128 * LDSW];

  const int tid   = threadIdx.x;
  const int lane  = tid & 31;
  const int wave  = tid >> 5;
  const int waveM = wave >> 2;          // 0..1
  const int waveN = wave & 3;           // 0..3
  const int m0    = blockIdx.y * 128;
  const int n0    = blockIdx.x * 128;
  const int lrow  = lane & 15;
  const int kbA   = (lane >> 4) * 8;    // A-frag chunk base
  const int kbB   = (lane >> 4) * 8;    // B-frag chunk base

  // stage one 128x32 bf16 slab of A and Bt into LDS buffer `buf`
  auto stage = [&](int buf, int kk) {
#pragma unroll
    for (int i = 0; i < 2; ++i) {
      int lin = tid + i * 256;          // 0..511 : 128 rows x 4 b128-chunks
      int row = lin >> 2;
      int ch  = (lin & 3) * 8;
      const __bf16* ga = &A[(size_t)(m0 + row) * K + kk + ch];
      const __bf16* gb = &Bt[(size_t)(n0 + row) * K + kk + ch];
      __bf16* la = &As[buf][row * LDSW + ch];
      __bf16* lb = &Bs[buf][row * LDSW + ch];
#if GEMM_ASYNC
      __builtin_amdgcn_global_load_async_to_lds_b128(
          (AS1 v4i_vs*)ga, (AS3 v4i_vs*)la, 0, 0);
      __builtin_amdgcn_global_load_async_to_lds_b128(
          (AS1 v4i_vs*)gb, (AS3 v4i_vs*)lb, 0, 0);
#else
      *(v8bf*)la = *(const v8bf*)ga;
      *(v8bf*)lb = *(const v8bf*)gb;
#endif
    }
  };

  v8f acc[4][2];
#pragma unroll
  for (int mi = 0; mi < 4; ++mi)
#pragma unroll
    for (int ni = 0; ni < 2; ++ni) acc[mi][ni] = zero8();

  stage(0, 0);
  wait_async_all();
  __syncthreads();

  for (int kk = 0; kk < K; kk += 32) {
    const int buf = (kk >> 5) & 1;
    if (kk + 32 < K) {
      stage(buf ^ 1, kk + 32);          // overlap next slab with this compute
#if !GEMM_ASYNC
      __builtin_prefetch(&A[(size_t)(m0 + (tid >> 1)) * K + kk + 64], 0, 1);
      __builtin_prefetch(&Bt[(size_t)(n0 + (tid >> 1)) * K + kk + 64], 0, 1);
#endif
    }

    const __bf16* Asb = &As[buf][0];
    const __bf16* Bsb = &Bs[buf][0];
    v16bf af[4], bfr[2];
#pragma unroll
    for (int mi = 0; mi < 4; ++mi) {
      int r = waveM * 64 + mi * 16 + lrow;
      af[mi] = make_frag(*(const v8bf*)(&Asb[r * LDSW + kbA]),
                         *(const v8bf*)(&Asb[r * LDSW + kbA + 16]));
    }
#pragma unroll
    for (int ni = 0; ni < 2; ++ni) {
      int r = waveN * 32 + ni * 16 + lrow;
      bfr[ni] = make_frag(*(const v8bf*)(&Bsb[r * LDSW + kbB]),
                          *(const v8bf*)(&Bsb[r * LDSW + kbB + 16]));
    }
#pragma unroll
    for (int mi = 0; mi < 4; ++mi)
#pragma unroll
      for (int ni = 0; ni < 2; ++ni)
        acc[mi][ni] = wmma_bf16(af[mi], bfr[ni], acc[mi][ni]);

    wait_async_all();                   // next slab landed in LDS
    __syncthreads();                    // one barrier per K-step (double buffer)
  }

  const int rbase = (lane >> 4) * 8;
#pragma unroll
  for (int mi = 0; mi < 4; ++mi)
#pragma unroll
    for (int ni = 0; ni < 2; ++ni)
#pragma unroll
      for (int r = 0; r < 8; ++r) {
        int row = m0 + waveM * 64 + mi * 16 + rbase + r;
        int col = n0 + waveN * 32 + ni * 16 + lrow;
        epi(row, col, acc[mi][ni][r]);
      }
}

// ------------------------------------------------------------ GEMM kernels ---

__global__ void __launch_bounds__(256)
qkv_gemm_kernel(const __bf16* __restrict__ X, const __bf16* __restrict__ WqkvT,
                const float* __restrict__ bias,
                __bf16* __restrict__ q, __bf16* __restrict__ k,
                __bf16* __restrict__ vT) {
  auto epi = [=](int row, int col, float v) {
    v += bias[col];
    __bf16 bv = f2bf(v);
    int sec = col >> 10;                 // 0:q 1:k 2:v
    int c   = col & 1023;
    int h   = c >> 6;
    int dd  = c & 63;
    int b   = row >> 11;
    int s   = row & 2047;
    size_t head = (size_t)(b * NH + h);
    if (sec == 0)      q[(head * SEQ + s) * HD + dd] = bv;
    else if (sec == 1) k[(head * SEQ + s) * HD + dd] = bv;
    else               vT[(head * HD + dd) * SEQ + s] = bv;   // V transposed
  };
  gemm_bf16_tile(X, WqkvT, HDIM, epi);
}

__global__ void __launch_bounds__(256)
out_gemm_kernel(const __bf16* __restrict__ A, const __bf16* __restrict__ WoutT,
                const float* __restrict__ bias, float* __restrict__ out) {
  auto epi = [=](int row, int col, float v) {
    out[(size_t)row * HDIM + col] = v + bias[col];
  };
  gemm_bf16_tile(A, WoutT, HDIM, epi);
}

// ------------------------------------------------- flash-attention kernel ---
// One wave owns a 16-row Q tile; streams causal K tiles of 32 columns.

__global__ void __launch_bounds__(256)
attn_kernel(const __bf16* __restrict__ q, const __bf16* __restrict__ kbuf,
            const __bf16* __restrict__ vT, const float* __restrict__ rel_bias,
            __bf16* __restrict__ attn) {
  __shared__ __bf16 Pl[8][16 * LDSW];   // per-wave P staging (f32 C -> bf16 A)

  const int tid   = threadIdx.x;
  const int lane  = tid & 31;
  const int wave  = tid >> 5;
  const int task  = blockIdx.x * 8 + wave;   // B*H*(S/16) tasks
  const int qt    = task & 127;
  const int h     = (task >> 7) & 15;
  const int b     = task >> 11;
  const int i0    = qt * 16;
  const int lrow  = lane & 15;
  const int kbA   = (lane >> 4) * 8;
  const int kbB   = (lane >> 4) * 8;
  const int rbase = (lane >> 4) * 8;

  const size_t head = (size_t)(b * NH + h);
  const __bf16* qp  = q    + (head * SEQ + i0) * HD;
  const __bf16* kp0 = kbuf + head * SEQ * HD;
  const __bf16* vp0 = vT   + head * (size_t)HD * SEQ;
  const float*  bp  = rel_bias + (size_t)h * SEQ * SEQ;

  // Q A-fragments for d = 0..63 (two K=32 steps), resident for whole row
  v16bf aq[2];
#pragma unroll
  for (int ks = 0; ks < 2; ++ks)
    aq[ks] = make_frag(*(const v8bf*)(qp + lrow * HD + ks * 32 + kbA),
                       *(const v8bf*)(qp + lrow * HD + ks * 32 + kbA + 16));

  float mrow[8], lsum[8], alpha[8];
#pragma unroll
  for (int r = 0; r < 8; ++r) { mrow[r] = -3.0e38f; lsum[r] = 0.f; }
  v8f o[4];
#pragma unroll
  for (int n = 0; n < 4; ++n) o[n] = zero8();

  __bf16* P = &Pl[wave][0];

  for (int j0 = 0; j0 < i0 + 16; j0 += 32) {
    // ---- S = Q K^T (two 16x16 column halves) --------------------------------
    v8f s[2]; s[0] = zero8(); s[1] = zero8();
#pragma unroll
    for (int t = 0; t < 2; ++t) {
      const __bf16* kp = kp0 + (size_t)(j0 + t * 16 + lrow) * HD;
#pragma unroll
      for (int ks = 0; ks < 2; ++ks) {
        v16bf bk = make_frag(*(const v8bf*)(kp + ks * 32 + kbB),
                             *(const v8bf*)(kp + ks * 32 + kbB + 16));
        s[t] = wmma_bf16(aq[ks], bk, s[t]);
      }
    }
    // ---- scale + rel_bias + causal mask ------------------------------------
    float p[2][8], mt[8];
#pragma unroll
    for (int r = 0; r < 8; ++r) {
      int i = i0 + rbase + r;
#pragma unroll
      for (int t = 0; t < 2; ++t) {
        int j = j0 + t * 16 + lrow;
        p[t][r] = (j <= i) ? (s[t][r] * 0.125f + bp[(size_t)i * SEQ + j])
                           : -3.0e38f;
      }
      mt[r] = fmaxf(p[0][r], p[1][r]);
    }
    // row max across the 16 lanes that hold one row (xor 1,2,4,8 stays in-half)
#pragma unroll
    for (int d = 1; d < 16; d <<= 1)
#pragma unroll
      for (int r = 0; r < 8; ++r)
        mt[r] = fmaxf(mt[r], __shfl_xor(mt[r], d, 32));
#pragma unroll
    for (int r = 0; r < 8; ++r) {
      float mn = fmaxf(mrow[r], mt[r]);
      alpha[r] = __expf(mrow[r] - mn);
      mrow[r]  = mn;
    }
    float psum[8];
#pragma unroll
    for (int r = 0; r < 8; ++r) {
#pragma unroll
      for (int t = 0; t < 2; ++t) p[t][r] = __expf(p[t][r] - mrow[r]);
      psum[r] = p[0][r] + p[1][r];
    }
#pragma unroll
    for (int d = 1; d < 16; d <<= 1)
#pragma unroll
      for (int r = 0; r < 8; ++r)
        psum[r] += __shfl_xor(psum[r], d, 32);
#pragma unroll
    for (int r = 0; r < 8; ++r) lsum[r] = lsum[r] * alpha[r] + psum[r];

    // ---- restage P (C-layout f32) as bf16 A-fragment via LDS ----------------
#pragma unroll
    for (int t = 0; t < 2; ++t)
#pragma unroll
      for (int r = 0; r < 8; ++r)
        P[(rbase + r) * LDSW + t * 16 + lrow] = f2bf(p[t][r]);
    asm volatile("s_wait_dscnt 0x0" ::: "memory");
    v16bf pf = make_frag(*(const v8bf*)(&P[lrow * LDSW + kbA]),
                         *(const v8bf*)(&P[lrow * LDSW + kbA + 16]));

    // ---- O = O*alpha + P V (four dd sub-tiles, K=32 over j) -----------------
#pragma unroll
    for (int n = 0; n < 4; ++n) {
#pragma unroll
      for (int r = 0; r < 8; ++r) o[n][r] *= alpha[r];
      const __bf16* vp = vp0 + (size_t)(n * 16 + lrow) * SEQ + j0;
      v16bf bv = make_frag(*(const v8bf*)(vp + kbB),
                           *(const v8bf*)(vp + kbB + 16));
      o[n] = wmma_bf16(pf, bv, o[n]);
    }
  }

  // ---- normalize and emit bf16 tokens [B*S][HDIM] for the output GEMM ------
#pragma unroll
  for (int r = 0; r < 8; ++r) {
    float inv = 1.0f / lsum[r];
    int row = b * SEQ + i0 + rbase + r;
#pragma unroll
    for (int n = 0; n < 4; ++n)
      attn[(size_t)row * HDIM + h * HD + n * 16 + lrow] = f2bf(o[n][r] * inv);
  }
}

// -------------------------------------------------------------- launcher ----

extern "C" void kernel_launch(void* const* d_in, const int* in_sizes, int n_in,
                              void* d_out, int out_size, void* d_ws, size_t ws_size,
                              hipStream_t stream) {
  const float* x    = (const float*)d_in[0];
  const float* rb   = (const float*)d_in[1];
  const float* wqkv = (const float*)d_in[2];
  const float* bqkv = (const float*)d_in[3];
  const float* wout = (const float*)d_in[4];
  const float* bout = (const float*)d_in[5];
  float* out = (float*)d_out;
  (void)in_sizes; (void)n_in; (void)out_size; (void)ws_size;

  char* ws = (char*)d_ws;
  size_t off = 0;
  auto carve = [&](size_t bytes) -> void* {
    void* p = ws + off;
    off += (bytes + 4095) & ~(size_t)4095;   // 4K pad between regions
    return p;
  };
  __bf16* xbf   = (__bf16*)carve((size_t)NTOK * HDIM * sizeof(__bf16));
  __bf16* wqkvT = (__bf16*)carve((size_t)3 * HDIM * HDIM * sizeof(__bf16));
  __bf16* woutT = (__bf16*)carve((size_t)HDIM * HDIM * sizeof(__bf16));
  __bf16* qb    = (__bf16*)carve((size_t)NTOK * HDIM * sizeof(__bf16));
  __bf16* kb    = (__bf16*)carve((size_t)NTOK * HDIM * sizeof(__bf16));
  __bf16* vTb   = (__bf16*)carve((size_t)NTOK * HDIM * sizeof(__bf16) + 8192);
  __bf16* attnb = (__bf16*)carve((size_t)NTOK * HDIM * sizeof(__bf16));

  cast_bf16_kernel<<<dim3(NTOK * HDIM / (256 * 4)), dim3(256), 0, stream>>>(
      x, xbf, NTOK * HDIM);
  transpose_cast_kernel<<<dim3(3 * HDIM / 32, HDIM / 32), dim3(32, 8), 0, stream>>>(
      wqkv, wqkvT, HDIM, 3 * HDIM);
  transpose_cast_kernel<<<dim3(HDIM / 32, HDIM / 32), dim3(32, 8), 0, stream>>>(
      wout, woutT, HDIM, HDIM);
  qkv_gemm_kernel<<<dim3(3 * HDIM / 128, NTOK / 128), dim3(256), 0, stream>>>(
      xbf, wqkvT, bqkv, qb, kb, vTb);
  attn_kernel<<<dim3(BATCH * NH * (SEQ / 16) / 8), dim3(256), 0, stream>>>(
      qb, kb, vTb, rb, attnb);
  out_gemm_kernel<<<dim3(HDIM / 128, NTOK / 128), dim3(256), 0, stream>>>(
      attnb, woutT, bout, out);
}
